// STSPBlock_54056458387721
// MI455X (gfx1250) — matrix operand order repaired and checked by reading, not compiled, not run on hardware
//
#include <hip/hip_runtime.h>
#include <hip/hip_bf16.h>

// ---------------- problem constants ----------------
#define T_STEPS 4
#define B_SZ    16
#define CIN     3
#define HIMG    64
#define WIMG    64
#define N_NODES 6
#define COUT    128
#define HPOOL   32
#define WPOOL   32
#define NUM_HEADS 4
#define HEAD_DIM  32
#define K_PRUNE 3
#define TRACE_DECAY 0.6f
#define BN_EPS 1e-5f
#define V_TH   1.0f

#define PIX0   (B_SZ*COUT*HIMG*WIMG)        // 8388608  conv0 output elems
#define PIXN   (B_SZ*COUT*HPOOL*WPOOL)      // 2097152  node conv elems
#define KMAIN  (COUT*9)                      // 1152

typedef __attribute__((ext_vector_type(16))) __bf16 v16bf;
typedef __attribute__((ext_vector_type(8)))  float  v8f;
typedef __attribute__((ext_vector_type(4)))  unsigned int v4u;
typedef __attribute__((ext_vector_type(8)))  int v8i;
typedef __attribute__((ext_vector_type(4)))  int v4i;

__device__ __forceinline__ float sigmoidf_(float x) { return 1.0f / (1.0f + __expf(-x)); }

__device__ __forceinline__ v8f wmma_bf16(v16bf a, v16bf b, v8f c) {
  // (neg_a, A, neg_b, B, c_mod, C, reuse_a, reuse_b)
  return __builtin_amdgcn_wmma_f32_16x16x32_bf16(false, a, false, b, (short)0, c, false, false);
}

// A fragment: 16x32 bf16 tile, row-major in LDS (stride 32). lane = hi*16 + m.
// VGPR element j -> K = hi*8+j (j<8), 16+hi*8+(j-8) (j>=8)   [ISA 7.12.2]
__device__ __forceinline__ v16bf load_a_frag(const __bf16* row, int hi) {
  v16bf a;
#pragma unroll
  for (int j = 0; j < 8; ++j) a[j] = row[hi * 8 + j];
#pragma unroll
  for (int j = 0; j < 8; ++j) a[8 + j] = row[16 + hi * 8 + j];
  return a;
}
// B fragment: 32x16 bf16 (KxN) from LDS row-major [32][ldb]; lane -> N (n0+lane%16),
// element j -> K = hi*8+j (j<8), 16+hi*8+(j-8) (j>=8).
__device__ __forceinline__ v16bf load_b_frag(const __bf16* Bs, int ldb, int lane, int n0) {
  v16bf b;
  int hi = lane >> 4;
  const __bf16* p = Bs + n0 + (lane & 15);
#pragma unroll
  for (int j = 0; j < 8; ++j) b[j] = p[(hi * 8 + j) * ldb];
#pragma unroll
  for (int j = 0; j < 8; ++j) b[8 + j] = p[(16 + hi * 8 + j) * ldb];
  return b;
}

// ---------------- Tensor Data Mover: async 2-D strided tile -> LDS ----------------
// D# per CDNA5 ISA ch.8: group0 {count=1, lds_addr, global_addr, type=2},
// group1 {wg_mask=0, data_size=1 (2B), dims/tile/stride}. Groups 2/3 zero (2-D).
// dims/tiles in elements (2B units); stride0 in elements.
__device__ __forceinline__ void tdm_load_2d_bf16(unsigned lds_off, const void* gaddr,
                                                 unsigned dim0, unsigned dim1,
                                                 unsigned tile0, unsigned tile1,
                                                 unsigned stride0) {
  unsigned long long ga = (unsigned long long)(size_t)gaddr;
  v4u g0;
  g0[0] = 1u;                                           // count=1
  g0[1] = lds_off;                                      // lds_addr (bytes)
  g0[2] = (unsigned)(ga & 0xFFFFFFFFu);                 // global_addr[31:0]
  g0[3] = (unsigned)((ga >> 32) & 0x01FFFFFFu) | (2u << 30);  // addr[56:32] | type=2
  v8i g1;
  g1[0] = (int)(1u << 16);                              // wg_mask=0, data_size=1 (2 bytes)
  g1[1] = (int)((dim0 & 0xFFFFu) << 16);                // tensor_dim0[15:0]
  g1[2] = (int)((dim0 >> 16) | ((dim1 & 0xFFFFu) << 16));     // dim0[31:16]|dim1[15:0]
  g1[3] = (int)((dim1 >> 16) | ((tile0 & 0xFFFFu) << 16));    // dim1[31:16]|tile_dim0
  g1[4] = (int)(tile1 & 0xFFFFu);                       // tile_dim1 (tile_dim2=0)
  g1[5] = (int)stride0;                                 // tensor_dim0_stride[31:0]
  g1[6] = 0;
  g1[7] = 0;
  v4i z4 = {0, 0, 0, 0};
#if defined(__clang_major__) && (__clang_major__ >= 23)
  v8i z8 = {0, 0, 0, 0, 0, 0, 0, 0};
  __builtin_amdgcn_tensor_load_to_lds(g0, g1, z4, z4, z8, 0);
#else
  __builtin_amdgcn_tensor_load_to_lds(g0, g1, z4, z4, 0);
#endif
}

// ---------------- weight/input conversion ----------------
__global__ void k_f32_to_bf16(const float* __restrict__ s, __bf16* __restrict__ d, int n) {
  int i = blockIdx.x * blockDim.x + threadIdx.x;
  if (i < n) d[i] = (__bf16)s[i];
}

// conv0 weights (128,3,3,3) -> [co][32] with K=27 padded to 32
__global__ void k_w0_pad(const float* __restrict__ w, __bf16* __restrict__ d) {
  int i = blockIdx.x * blockDim.x + threadIdx.x;
  if (i >= COUT * 32) return;
  int co = i >> 5, k = i & 31;
  d[i] = (k < 27) ? (__bf16)w[co * 27 + k] : (__bf16)0.0f;
}

// fold BN into per-channel scale/shift: slot 0 = bn0, slots 1..5 = bns[i-1]
__global__ void k_bnparams(const float* g0, const float* b0, const float* m0, const float* v0,
                           const float* gs, const float* bs, const float* ms, const float* vs,
                           float* __restrict__ scale, float* __restrict__ shift) {
  int i = blockIdx.x * blockDim.x + threadIdx.x;
  if (i >= N_NODES * COUT) return;
  int grp = i >> 7, c = i & 127;
  float g, b, m, v;
  if (grp == 0) { g = g0[c]; b = b0[c]; m = m0[c]; v = v0[c]; }
  else { int j = (grp - 1) * COUT + c; g = gs[j]; b = bs[j]; m = ms[j]; v = vs[j]; }
  float sc = g * rsqrtf(v + BN_EPS);
  scale[i] = sc;
  shift[i] = b - m * sc;
}

// ---------------- conv0: implicit GEMM, M=128, K=32(27), N = B*64*64 ----------------
// grid (1024, 2): blockIdx.x = b*64 + h (one full image row = 64 pixels), blockIdx.y = mblk
__global__ __launch_bounds__(128) void k_conv0(
    const __bf16* __restrict__ xb, const __bf16* __restrict__ w0,
    const float* __restrict__ bns, const float* __restrict__ bnb,
    float* __restrict__ xc) {
  __shared__ __bf16 As[64 * 32];
  __shared__ __bf16 Bs[32 * 64];
  int t = threadIdx.x;
  int nrow = blockIdx.x, mblk = blockIdx.y;
  int b = nrow >> 6, h = nrow & 63;
  { // stage A: 64x32 weights, contiguous
    const uint4* s = (const uint4*)(w0 + mblk * 64 * 32);
    uint4* d = (uint4*)As;
    d[t] = s[t]; d[t + 128] = s[t + 128];
  }
  { // stage B: im2col row, k -> (ci,kh,kw)
    int r = t >> 2, seg = t & 3;
    int ci = r / 9; if (ci > CIN - 1) ci = CIN - 1;
    int rr = r - (r / 9) * 9;
    int kh = rr / 3, kw = rr - kh * 3;
    int y = h + kh - 1;
    bool okk = (r < 27) && (y >= 0) && (y < HIMG);
    const __bf16* src = xb + ((b * CIN + ci) * HIMG + (okk ? y : 0)) * WIMG;
#pragma unroll
    for (int j = 0; j < 16; ++j) {
      int n = seg * 16 + j;
      int x = n + kw - 1;
      __bf16 v = (__bf16)0.0f;
      if (okk && x >= 0 && x < WIMG) v = src[x];
      Bs[r * 64 + n] = v;
    }
  }
  __syncthreads();
  int lane = t & 31, wid = t >> 5;
  int wm = (wid >> 1) * 32, wn = (wid & 1) * 32;
  int hi = lane >> 4, mr = lane & 15, nl = lane & 15;
  v8f acc[2][2] = {};
  v16bf a0 = load_a_frag(As + (wm + mr) * 32, hi);
  v16bf a1 = load_a_frag(As + (wm + 16 + mr) * 32, hi);
  v16bf bf0 = load_b_frag(Bs, 64, lane, wn);
  v16bf bf1 = load_b_frag(Bs, 64, lane, wn + 16);
  acc[0][0] = wmma_bf16(a0, bf0, acc[0][0]);
  acc[0][1] = wmma_bf16(a0, bf1, acc[0][1]);
  acc[1][0] = wmma_bf16(a1, bf0, acc[1][0]);
  acc[1][1] = wmma_bf16(a1, bf1, acc[1][1]);
#pragma unroll
  for (int tm = 0; tm < 2; ++tm)
#pragma unroll
    for (int tn = 0; tn < 2; ++tn)
#pragma unroll
      for (int r = 0; r < 8; ++r) {
        int co = mblk * 64 + wm + tm * 16 + r + 8 * hi;
        int w = wn + tn * 16 + nl;
        xc[((b * COUT + co) * HIMG + h) * WIMG + w] = acc[tm][tn][r] * bns[co] + bnb[co];
      }
}

// ---------------- node conv: implicit GEMM, M=128, K=1152, N = B*32*32 ----------------
// A tile (weights, strided 2-D) is DMA'd into LDS by the Tensor Data Mover while the
// wave stages the im2col B tile; synchronized via TENSORcnt + workgroup barrier.
// grid (1024, 2): blockIdx.x = 64-pixel tile (2 rows of one image), blockIdx.y = mblk
__global__ __launch_bounds__(128) void k_conv_main(
    const __bf16* __restrict__ inb, const __bf16* __restrict__ wb,
    const float* __restrict__ bns, const float* __restrict__ bnb,
    float* __restrict__ outp) {
  __shared__ __bf16 As[64 * 32];
  __shared__ __bf16 Bs[32 * 64];
  int t = threadIdx.x;
  int nblk = blockIdx.x, mblk = blockIdx.y;
  int p0 = nblk * 64;
  int b = p0 >> 10;
  int h0 = (p0 & 1023) >> 5;
  int lane = t & 31, wid = t >> 5;
  int wm = (wid >> 1) * 32, wn = (wid & 1) * 32;
  int hi = lane >> 4, mr = lane & 15, nl = lane & 15;
  int brow = t >> 2, bseg = t & 3;
  unsigned as_off = (unsigned)(size_t)(&As[0]);   // LDS byte offset (low 32 bits)
  const __bf16* wbase = wb + (size_t)(mblk * 64) * KMAIN;
  v8f acc[2][2] = {};
  for (int kt = 0; kt < KMAIN / 32; ++kt) {
    __syncthreads();
    if (t < 32) {
      // async DMA: 64 rows x 32 cols of bf16 weights, row stride KMAIN elements
      tdm_load_2d_bf16(as_off, wbase + kt * 32, 32, 64, 32, 64, KMAIN);
    }
    { // stage B: im2col (gathered, zero-padded) -> LDS
      int k = kt * 32 + brow;
      int ci = k / 9;
      int rr = k - ci * 9;
      int kh = rr / 3, kw = rr - kh * 3;
      const __bf16* base = inb + (size_t)(b * COUT + ci) * HPOOL * WPOOL;
#pragma unroll
      for (int j = 0; j < 16; ++j) {
        int n = bseg * 16 + j;
        int h = h0 + (n >> 5), w = n & 31;
        int y = h + kh - 1, x = w + kw - 1;
        __bf16 v = (__bf16)0.0f;
        if (y >= 0 && y < HPOOL && x >= 0 && x < WPOOL) v = base[y * WPOOL + x];
        Bs[brow * 64 + n] = v;
      }
    }
    if (t < 32) __builtin_amdgcn_s_wait_tensorcnt(0);
    __syncthreads();
    v16bf a0 = load_a_frag(As + (wm + mr) * 32, hi);
    v16bf a1 = load_a_frag(As + (wm + 16 + mr) * 32, hi);
    v16bf bf0 = load_b_frag(Bs, 64, lane, wn);
    v16bf bf1 = load_b_frag(Bs, 64, lane, wn + 16);
    acc[0][0] = wmma_bf16(a0, bf0, acc[0][0]);
    acc[0][1] = wmma_bf16(a0, bf1, acc[0][1]);
    acc[1][0] = wmma_bf16(a1, bf0, acc[1][0]);
    acc[1][1] = wmma_bf16(a1, bf1, acc[1][1]);
  }
#pragma unroll
  for (int tm = 0; tm < 2; ++tm)
#pragma unroll
    for (int tn = 0; tn < 2; ++tn)
#pragma unroll
      for (int r = 0; r < 8; ++r) {
        int co = mblk * 64 + wm + tm * 16 + r + 8 * hi;
        int n = wn + tn * 16 + nl;
        int h = h0 + (n >> 5), w = n & 31;
        outp[((b * COUT + co) * HPOOL + h) * WPOOL + w] = acc[tm][tn][r] * bns[co] + bnb[co];
      }
}

// ---------------- generic 1-wave WMMA GEMM: C = act(A[M,K] * W[N,K]^T + bias) ----------------
__global__ __launch_bounds__(32) void k_gemm_bt(
    const float* __restrict__ A, const float* __restrict__ W,
    const float* __restrict__ bias, float* __restrict__ C,
    int K, int ldc, int cOff, int doRelu) {
  __shared__ __bf16 As[16 * 32];
  __shared__ __bf16 Bs[32 * 16];
  int t = threadIdx.x;
  int m0 = blockIdx.x * 16, n0 = blockIdx.y * 16;
  int hi = t >> 4, mr = t & 15, nl = t & 15;
  v8f acc = {};
  for (int k0 = 0; k0 < K; k0 += 32) {
    __syncthreads();
    for (int idx = t; idx < 512; idx += 32) {
      int m = idx >> 5, kk = idx & 31;
      As[idx] = (__bf16)A[(m0 + m) * K + k0 + kk];
    }
    for (int idx = t; idx < 512; idx += 32) {
      int k = idx >> 4, n = idx & 15;
      Bs[idx] = (__bf16)W[(n0 + n) * K + k0 + k];
    }
    __syncthreads();
    v16bf a = load_a_frag(As + mr * 32, hi);
    v16bf b = load_b_frag(Bs, 16, t, 0);
    acc = wmma_bf16(a, b, acc);
  }
#pragma unroll
  for (int r = 0; r < 8; ++r) {
    int m = m0 + r + 8 * hi;
    int n = n0 + nl;
    float v = acc[r];
    if (bias) v += bias[n];
    if (doRelu) v = v > 0.f ? v : 0.f;
    C[cOff + m * ldc + n] = v;
  }
}

// ---------------- PLIF + 2x2 avg-pool for node 0; also base-writes out[t] ----------------
__global__ void k_plif_pool0(const float* __restrict__ xc, float* __restrict__ v0,
                             const float* __restrict__ plif_w, const float* __restrict__ out_w,
                             float* __restrict__ out0, float* __restrict__ outT) {
  int i = blockIdx.x * blockDim.x + threadIdx.x;
  if (i >= PIXN) return;
  int pw = i & 31, ph = (i >> 5) & 31, bc = i >> 10;
  float sig = sigmoidf_(plif_w[0]);
  float s = 0.f;
#pragma unroll
  for (int dy = 0; dy < 2; ++dy)
#pragma unroll
    for (int dx = 0; dx < 2; ++dx) {
      int idx = (bc * HIMG + 2 * ph + dy) * WIMG + 2 * pw + dx;
      float v = v0[idx];
      v = v + (xc[idx] - v) * sig;
      float sp = (v >= V_TH) ? 1.f : 0.f;
      v0[idx] = v * (1.f - sp);
      s += sp;
    }
  float o = s * 0.25f;
  out0[i] = o;
  outT[i] = sigmoidf_(out_w[0]) * o;   // node 0 contribution (base write)
}

// ---------------- PLIF for node i; spike out + weighted accumulate into out[t] ----------------
__global__ void k_plif_node(const float* __restrict__ ci, float* __restrict__ vst,
                            const float* __restrict__ plif_w, int node,
                            const float* __restrict__ out_w,
                            float* __restrict__ spike, float* __restrict__ outT) {
  int i = blockIdx.x * blockDim.x + threadIdx.x;
  if (i >= PIXN) return;
  float sig = sigmoidf_(plif_w[node]);
  float v = vst[i];
  v = v + (ci[i] - v) * sig;
  float sp = (v >= V_TH) ? 1.f : 0.f;
  vst[i] = v * (1.f - sp);
  spike[i] = sp;
  outT[i] += sigmoidf_(out_w[node]) * sp;
}

// ---------------- spatial mean over 32x32 per (b,c) ----------------
__global__ __launch_bounds__(128) void k_rowmean(const float* __restrict__ in, float* __restrict__ out) {
  __shared__ float red[128];
  int blk = blockIdx.x, t = threadIdx.x;
  float s = 0.f;
  for (int j = t; j < 1024; j += 128) s += in[blk * 1024 + j];
  red[t] = s; __syncthreads();
  for (int st = 64; st > 0; st >>= 1) { if (t < st) red[t] += red[t + st]; __syncthreads(); }
  if (t == 0) out[blk] = red[0] * (1.f / 1024.f);
}

// traces[:,0,:] = decay*traces[:,0,:] + (1-decay)*feats[:,0,:]
__global__ void k_trace_row0(float* __restrict__ traces, const float* __restrict__ feats) {
  int i = blockIdx.x * blockDim.x + threadIdx.x;
  if (i >= B_SZ * COUT) return;
  int b = i >> 7, c = i & 127;
  int idx = b * N_NODES * COUT + c;
  traces[idx] = TRACE_DECAY * traces[idx] + (1.f - TRACE_DECAY) * feats[idx];
}

// traces = decay*traces + (1-decay)*feats (all nodes)
__global__ void k_traces_all(float* __restrict__ traces, const float* __restrict__ feats) {
  int i = blockIdx.x * blockDim.x + threadIdx.x;
  if (i >= B_SZ * N_NODES * COUT) return;
  traces[i] = TRACE_DECAY * traces[i] + (1.f - TRACE_DECAY) * feats[i];
}

// ---------------- GAT attention + topk prune + diffusion^2 coefficients ----------------
// one wave per batch; Xs has only node-0 nonzero, so Hd[:,i,:] = op2[b,i,0]*out0
__global__ __launch_bounds__(32) void k_gat(const float* __restrict__ hp,
                                            const float* __restrict__ ga,
                                            float* __restrict__ coef) {
  int b = blockIdx.x, l = threadIdx.x;
  float e1v[N_NODES][NUM_HEADS], e2v[N_NODES][NUM_HEADS];
#pragma unroll
  for (int n = 0; n < N_NODES; ++n)
#pragma unroll
    for (int h = 0; h < NUM_HEADS; ++h) {
      float v = hp[(b * N_NODES + n) * COUT + h * HEAD_DIM + l];
      float s1 = v * ga[h * 2 * HEAD_DIM + l];
      float s2 = v * ga[h * 2 * HEAD_DIM + HEAD_DIM + l];
#pragma unroll
      for (int off = 16; off > 0; off >>= 1) {
        s1 += __shfl_xor(s1, off, 32);
        s2 += __shfl_xor(s2, off, 32);
      }
      e1v[n][h] = s1; e2v[n][h] = s2;
    }
  if (l == 0) {
    float S[N_NODES][N_NODES];
    for (int i = 0; i < N_NODES; ++i) {
      float row[N_NODES];
      for (int j = 0; j < N_NODES; ++j) {
        float s = 0.f;
        for (int h = 0; h < NUM_HEADS; ++h) {
          float e = e1v[i][h] + e2v[j][h];   e = (e > 0.f) ? e : 0.2f * e;
          float et = e1v[j][h] + e2v[i][h];  et = (et > 0.f) ? et : 0.2f * et;
          s += 0.5f * (e + et);
        }
        row[j] = s * (1.f / NUM_HEADS);
      }
      float mx = row[0];
      for (int j = 1; j < N_NODES; ++j) mx = fmaxf(mx, row[j]);
      float sm = 0.f;
      for (int j = 0; j < N_NODES; ++j) { row[j] = __expf((row[j] - mx) * 100.f); sm += row[j]; }
      for (int j = 0; j < N_NODES; ++j) S[i][j] = row[j] / sm;
    }
    // prune: keep entries >= 3rd largest per row
    for (int i = 0; i < N_NODES; ++i) {
      bool used[N_NODES]; float kth = 0.f;
      for (int j = 0; j < N_NODES; ++j) used[j] = false;
      for (int p = 0; p < K_PRUNE; ++p) {
        int bi = 0; float bv = -1e30f;
        for (int j = 0; j < N_NODES; ++j)
          if (!used[j] && S[i][j] > bv) { bv = S[i][j]; bi = j; }
        used[bi] = true; kth = bv;
      }
      for (int j = 0; j < N_NODES; ++j) S[i][j] = (S[i][j] >= kth) ? S[i][j] : 0.f;
    }
    // symmetrize + degree normalize
    float A2[N_NODES][N_NODES], dis[N_NODES];
    for (int i = 0; i < N_NODES; ++i) {
      float d = 0.f;
      for (int j = 0; j < N_NODES; ++j) { A2[i][j] = 0.5f * (S[i][j] + S[j][i]); d += A2[i][j]; }
      dis[i] = rsqrtf(d + 1e-6f);
    }
    float op[N_NODES][N_NODES];
    for (int i = 0; i < N_NODES; ++i)
      for (int j = 0; j < N_NODES; ++j)
        op[i][j] = dis[i] * A2[i][j] * dis[j];
    for (int i = 0; i < N_NODES; ++i) {
      float c = 0.f;
      for (int j = 0; j < N_NODES; ++j) c += op[i][j] * op[j][0];
      coef[b * N_NODES + i] = c;
    }
  }
}

// in_i (bf16) = coef[b,i] * out0
__global__ void k_scale_bf16(const float* __restrict__ out0, const float* __restrict__ coef,
                             int node, __bf16* __restrict__ dst) {
  int i = blockIdx.x * blockDim.x + threadIdx.x;
  if (i >= PIXN) return;
  int b = i >> 17;                       // COUT*HPOOL*WPOOL = 131072
  dst[i] = (__bf16)(coef[b * N_NODES + node] * out0[i]);
}

// ------------------------------------------------------------------
extern "C" void kernel_launch(void* const* d_in, const int* in_sizes, int n_in,
                              void* d_out, int out_size, void* d_ws, size_t ws_size,
                              hipStream_t stream) {
  (void)in_sizes; (void)n_in; (void)out_size; (void)ws_size;
  const float* x       = (const float*)d_in[0];
  const float* conv0_w = (const float*)d_in[1];
  const float* bn0_g   = (const float*)d_in[2];
  const float* bn0_b   = (const float*)d_in[3];
  const float* bn0_m   = (const float*)d_in[4];
  const float* bn0_v   = (const float*)d_in[5];
  const float* convs_w = (const float*)d_in[6];
  const float* bns_g   = (const float*)d_in[7];
  const float* bns_b   = (const float*)d_in[8];
  const float* bns_m   = (const float*)d_in[9];
  const float* bns_v   = (const float*)d_in[10];
  const float* plif_w  = (const float*)d_in[11];
  const float* ft_w    = (const float*)d_in[12];
  const float* ft_b    = (const float*)d_in[13];
  const float* gat_W   = (const float*)d_in[14];
  const float* gat_a   = (const float*)d_in[15];
  const float* out_w   = (const float*)d_in[16];
  float* outp = (float*)d_out;

  char* ws = (char*)d_ws;
  size_t off = 0;
  auto alloc = [&](size_t bytes) -> void* {
    char* p = ws + off;
    off += (bytes + 255) & ~(size_t)255;
    return (void*)p;
  };

  const int NXBF  = T_STEPS * B_SZ * CIN * HIMG * WIMG;  // 786432
  const int NWSBF = (N_NODES - 1) * COUT * KMAIN;        // 737280

  __bf16* xbf     = (__bf16*)alloc(sizeof(__bf16) * NXBF);
  __bf16* w0bf    = (__bf16*)alloc(sizeof(__bf16) * COUT * 32);
  __bf16* wsbf    = (__bf16*)alloc(sizeof(__bf16) * NWSBF);
  float*  bnscale = (float*)alloc(sizeof(float) * N_NODES * COUT);
  float*  bnshift = (float*)alloc(sizeof(float) * N_NODES * COUT);
  float*  xc      = (float*)alloc(sizeof(float) * PIX0);
  float*  v0buf   = (float*)alloc(sizeof(float) * PIX0);
  float*  out0    = (float*)alloc(sizeof(float) * PIXN);
  float*  vnodes  = (float*)alloc(sizeof(float) * (N_NODES - 1) * PIXN);
  __bf16* inbf    = (__bf16*)alloc(sizeof(__bf16) * PIXN);
  float*  cibuf   = (float*)alloc(sizeof(float) * PIXN);
  float*  spikeb  = (float*)alloc(sizeof(float) * PIXN);
  float*  meanb   = (float*)alloc(sizeof(float) * B_SZ * COUT);
  float*  feats   = (float*)alloc(sizeof(float) * B_SZ * N_NODES * COUT);
  float*  traces  = (float*)alloc(sizeof(float) * B_SZ * N_NODES * COUT);
  float*  hpbuf   = (float*)alloc(sizeof(float) * B_SZ * N_NODES * COUT);
  float*  coefb   = (float*)alloc(sizeof(float) * B_SZ * N_NODES);

  // zero persistent state (graph-capturable)
  hipMemsetAsync(v0buf, 0, sizeof(float) * PIX0, stream);
  hipMemsetAsync(vnodes, 0, sizeof(float) * (N_NODES - 1) * PIXN, stream);
  hipMemsetAsync(traces, 0, sizeof(float) * B_SZ * N_NODES * COUT, stream);

  // one-time conversions
  k_f32_to_bf16<<<(NXBF + 255) / 256, 256, 0, stream>>>(x, xbf, NXBF);
  k_w0_pad<<<(COUT * 32 + 255) / 256, 256, 0, stream>>>(conv0_w, w0bf);
  k_f32_to_bf16<<<(NWSBF + 255) / 256, 256, 0, stream>>>(convs_w, wsbf, NWSBF);
  k_bnparams<<<3, 256, 0, stream>>>(bn0_g, bn0_b, bn0_m, bn0_v,
                                    bns_g, bns_b, bns_m, bns_v, bnscale, bnshift);

  for (int t = 0; t < T_STEPS; ++t) {
    const __bf16* xbt = xbf + (size_t)t * B_SZ * CIN * HIMG * WIMG;
    float* outT = outp + (size_t)t * PIXN;

    // node 0: conv + BN (fused) -> PLIF + pool -> mean -> feat FC
    k_conv0<<<dim3(B_SZ * HIMG, 2), 128, 0, stream>>>(xbt, w0bf, bnscale, bnshift, xc);
    k_plif_pool0<<<PIXN / 256, 256, 0, stream>>>(xc, v0buf, plif_w, out_w, out0, outT);
    k_rowmean<<<B_SZ * COUT, 128, 0, stream>>>(out0, meanb);
    k_gemm_bt<<<dim3(1, COUT / 16), 32, 0, stream>>>(meanb, ft_w, ft_b, feats,
                                                     COUT, N_NODES * COUT, 0, 1);
    k_trace_row0<<<(B_SZ * COUT + 255) / 256, 256, 0, stream>>>(traces, feats);

    // GAT: hp = traces @ gat_W^T, then per-batch attention -> diffusion^2 coefficients
    k_gemm_bt<<<dim3(B_SZ * N_NODES / 16, COUT / 16), 32, 0, stream>>>(
        traces, gat_W, nullptr, hpbuf, COUT, COUT, 0, 0);
    k_gat<<<B_SZ, 32, 0, stream>>>(hpbuf, gat_a, coefb);

    // nodes 1..5
    for (int i = 1; i < N_NODES; ++i) {
      k_scale_bf16<<<PIXN / 256, 256, 0, stream>>>(out0, coefb, i, inbf);
      k_conv_main<<<dim3(PIXN / COUT / 64 * 4, 2), 128, 0, stream>>>(
          inbf, wsbf + (size_t)(i - 1) * COUT * KMAIN,
          bnscale + i * COUT, bnshift + i * COUT, cibuf);
      k_plif_node<<<PIXN / 256, 256, 0, stream>>>(
          cibuf, vnodes + (size_t)(i - 1) * PIXN, plif_w, i, out_w, spikeb, outT);
      k_rowmean<<<B_SZ * COUT, 128, 0, stream>>>(spikeb, meanb);
      k_gemm_bt<<<dim3(1, COUT / 16), 32, 0, stream>>>(meanb, ft_w, ft_b, feats,
                                                       COUT, N_NODES * COUT, i * COUT, 1);
    }
    k_traces_all<<<(B_SZ * N_NODES * COUT + 255) / 256, 256, 0, stream>>>(traces, feats);
  }
}
